// SigmaMoE_24146306138174
// MI455X (gfx1250) — matrix operand order repaired and checked by reading, not compiled, not run on hardware
//
#include <hip/hip_runtime.h>
#include <hip/hip_bf16.h>

typedef __attribute__((ext_vector_type(16))) __bf16 v16bf;
typedef __attribute__((ext_vector_type(8)))  __bf16 v8bf;
typedef __attribute__((ext_vector_type(8)))  float  v8f;

#define TOK  8192      // B*S
#define DM   1024      // d_model
#define FE   512       // d_expert
#define NE   16        // experts
#define NR   14        // routed experts
#define KR   4         // top-k routed
#define KSEL 6         // routed + shared
#define MT   32        // tokens per workgroup tile (2x B-fragment reuse)

// fast sigmoid via v_rcp_f32 (single transcendental instead of IEEE divide)
__device__ __forceinline__ float fast_sigmoid(float x) {
    return __builtin_amdgcn_rcpf(1.f + __expf(-x));
}

// ---------------------------------------------------------------- init
__global__ __launch_bounds__(256) void moe_init(float* __restrict__ out,
                                                int* __restrict__ counts) {
    size_t i = (size_t)blockIdx.x * blockDim.x + threadIdx.x;
    size_t n = (size_t)TOK * DM;
    for (size_t j = i; j < n; j += (size_t)gridDim.x * blockDim.x) out[j] = 0.f;
    if (i < NE) counts[i] = 0;
}

// ------------------------------------------------- weight fp32 -> bf16
__global__ __launch_bounds__(256) void moe_cvt(const float* __restrict__ keys_w,
                                               const float* __restrict__ values_w,
                                               __bf16* __restrict__ keysB,
                                               __bf16* __restrict__ valuesB) {
    const size_t half = (size_t)NE * DM * FE;           // 8,388,608
    size_t i = (size_t)blockIdx.x * blockDim.x + threadIdx.x;
    for (size_t j = i; j < 2 * half; j += (size_t)gridDim.x * blockDim.x) {
        if (j < half) keysB[j]          = (__bf16)keys_w[j];
        else          valuesB[j - half] = (__bf16)values_w[j - half];
    }
}

// ---------------------------------------------------------------- routing
// one wave32 per token: 16 dots of length 1024, sigmoid, top-4(+bias), shared
__global__ __launch_bounds__(256) void moe_route(const float* __restrict__ sel_in,
                                                 const float* __restrict__ es,
                                                 const float* __restrict__ bias,
                                                 float* __restrict__ wdense,
                                                 int* __restrict__ sel_out,
                                                 int* __restrict__ counts) {
    int wave = threadIdx.x >> 5;
    int lane = threadIdx.x & 31;
    int t = blockIdx.x * 8 + wave;
    if (t >= TOK) return;

    float x[32];
    const float* xp = sel_in + (size_t)t * DM;
#pragma unroll
    for (int i = 0; i < 32; ++i) x[i] = xp[i * 32 + lane];

    float acc[NE];
#pragma unroll
    for (int e = 0; e < NE; ++e) acc[e] = 0.f;
    for (int e = 0; e < NE; ++e) {
        const float* ep = es + (size_t)e * DM;
#pragma unroll
        for (int i = 0; i < 32; ++i) acc[e] += x[i] * ep[i * 32 + lane];
    }
#pragma unroll
    for (int e = 0; e < NE; ++e)
#pragma unroll
        for (int off = 16; off > 0; off >>= 1)
            acc[e] += __shfl_xor(acc[e], off, 32);

    if (lane == 0) {
        float aff[NE];
#pragma unroll
        for (int e = 0; e < NE; ++e) aff[e] = fast_sigmoid(acc[e]);
        float rank[NR];
#pragma unroll
        for (int e = 0; e < NR; ++e) rank[e] = aff[e] + bias[e];
        int chosen[KSEL];
        for (int k = 0; k < KR; ++k) {
            int best = 0; float bv = -3.0e38f;
#pragma unroll
            for (int e = 0; e < NR; ++e)
                if (rank[e] > bv) { bv = rank[e]; best = e; }
            chosen[k] = best;
            rank[best] = -3.0e38f;
        }
        chosen[4] = NR; chosen[5] = NR + 1;   // shared experts 14, 15
        float* wrow = wdense + (size_t)t * NE;
#pragma unroll
        for (int e = 0; e < NE; ++e) wrow[e] = 0.f;
        for (int k = 0; k < KSEL; ++k) {
            wrow[chosen[k]] = aff[chosen[k]];
            sel_out[t * KSEL + k] = chosen[k];
            atomicAdd(&counts[chosen[k]], 1);
        }
    }
}

// ---------------------------------------------------------------- scan
__global__ void moe_scan(const int* __restrict__ counts,
                         int* __restrict__ offsets, int* __restrict__ cursor) {
    if (threadIdx.x == 0 && blockIdx.x == 0) {
        int acc = 0;
        for (int e = 0; e < NE; ++e) { offsets[e] = acc; cursor[e] = acc; acc += counts[e]; }
    }
}

// ---------------------------------------------------------------- scatter
__global__ __launch_bounds__(256) void moe_scatter(const int* __restrict__ sel_out,
                                                   int* __restrict__ cursor,
                                                   int* __restrict__ pair_token) {
    int t = blockIdx.x * blockDim.x + threadIdx.x;
    if (t >= TOK) return;
    for (int k = 0; k < KSEL; ++k) {
        int e = sel_out[t * KSEL + k];
        int pos = atomicAdd(&cursor[e], 1);
        pair_token[pos] = t;
    }
}

// ------------------------------------------------- grouped WMMA expert FFN
// block = 8 waves; blockIdx.y = expert, blockIdx.x = 32-token tile of that
// expert's bucket.  Each B-fragment (32K x 16N, 1KB) feeds TWO wmma ops
// (M rows 0-15 and 16-31) -> 2x arithmetic intensity vs a 16-row tile.
// Stage 1: S = silu(X @ K_e) * aff (32x512, K=1024).
// Stage 2: out += S @ V_e (32x1024, K=512).
__global__ __launch_bounds__(256) void moe_gemm(const float* __restrict__ token_stream,
                                                const __bf16* __restrict__ keysB,
                                                const __bf16* __restrict__ valuesB,
                                                const float* __restrict__ wdense,
                                                const int* __restrict__ pair_token,
                                                const int* __restrict__ counts,
                                                const int* __restrict__ offsets,
                                                float* __restrict__ out) {
    const int e    = blockIdx.y;
    const int tile = blockIdx.x;
    const int cnt  = counts[e];
    if (tile * MT >= cnt) return;
    const int base = offsets[e] + tile * MT;

    __shared__ __bf16 Xt[MT][DM + 8];   // 66 KB
    __shared__ __bf16 St[MT][FE + 8];   // 33 KB
    __shared__ int    toks[MT];
    __shared__ float  affr[MT];

    const int tid = threadIdx.x;
    if (tid < MT) {
        bool valid = (tile * MT + tid) < cnt;
        int tok = valid ? pair_token[base + tid] : 0;
        toks[tid] = tok;
        affr[tid] = valid ? wdense[(size_t)tok * NE + e] : 0.f;  // 0-weight pad rows
    }
    __syncthreads();

    // cooperative gather of the 32xD activation tile, fp32 -> bf16
    {
        int row = tid >> 3;            // 32 rows, 8 threads per row
        int cb  = (tid & 7) * 128;     // 128 cols per thread
        const float* src = token_stream + (size_t)toks[row] * DM + cb;
#pragma unroll
        for (int i = 0; i < 128; i += 4) {
            float4 v = *(const float4*)(src + i);
            Xt[row][cb + i + 0] = (__bf16)v.x;
            Xt[row][cb + i + 1] = (__bf16)v.y;
            Xt[row][cb + i + 2] = (__bf16)v.z;
            Xt[row][cb + i + 3] = (__bf16)v.w;
        }
    }
    __syncthreads();

    const int wave = tid >> 5;
    const int lane = tid & 31;
    const int m    = lane & 15;      // A row / C col index
    const int kh   = lane >> 4;      // K-half select

    // ---------------- stage 1: scores = silu(X @ K_e) * aff  -> St
    const __bf16* Ke = keysB + (size_t)e * DM * FE;
    for (int nt = wave; nt < FE / 16; nt += 8) {
        const int n0 = nt * 16;
        v8f c0 = {};
        v8f c1 = {};
        for (int d0 = 0; d0 < DM; d0 += 32) {
            const __bf16* bp = Ke + (size_t)(d0 + 8 * kh) * FE + n0 + m;
            __builtin_prefetch(bp + 32 * FE, 0, 1);
            v16bf b;
#pragma unroll
            for (int s = 0; s < 8; ++s) { b[s] = bp[(size_t)s * FE]; b[s + 8] = bp[(size_t)(16 + s) * FE]; }

            v8bf lo0 = *(const v8bf*)&Xt[m][d0 + 8 * kh];
            v8bf hi0 = *(const v8bf*)&Xt[m][d0 + 16 + 8 * kh];
            v16bf a0 = __builtin_shufflevector(lo0, hi0, 0, 1, 2, 3, 4, 5, 6, 7,
                                               8, 9, 10, 11, 12, 13, 14, 15);
            c0 = __builtin_amdgcn_wmma_f32_16x16x32_bf16(false, a0, false, b,
                                                         (short)0, c0, false, false);

            v8bf lo1 = *(const v8bf*)&Xt[m + 16][d0 + 8 * kh];
            v8bf hi1 = *(const v8bf*)&Xt[m + 16][d0 + 16 + 8 * kh];
            v16bf a1 = __builtin_shufflevector(lo1, hi1, 0, 1, 2, 3, 4, 5, 6, 7,
                                               8, 9, 10, 11, 12, 13, 14, 15);
            c1 = __builtin_amdgcn_wmma_f32_16x16x32_bf16(false, a1, false, b,
                                                         (short)0, c1, false, false);
        }
        const int f = n0 + m;        // C: N = lane&15
#pragma unroll
        for (int r = 0; r < 8; ++r) {
            int M = r + 8 * kh;      // C: M = vgpr + 8*lane[4]
            float v0 = c0[r];
            St[M][f] = (__bf16)(v0 * fast_sigmoid(v0) * affr[M]);        // silu*aff
            float v1 = c1[r];
            St[M + 16][f] = (__bf16)(v1 * fast_sigmoid(v1) * affr[M + 16]);
        }
    }
    __syncthreads();

    // ---------------- stage 2: out += St @ V_e
    const __bf16* Ve = valuesB + (size_t)e * FE * DM;
    for (int nt = wave; nt < DM / 16; nt += 8) {
        const int n0 = nt * 16;
        v8f c0 = {};
        v8f c1 = {};
        for (int f0 = 0; f0 < FE; f0 += 32) {
            const __bf16* bp = Ve + (size_t)(f0 + 8 * kh) * DM + n0 + m;
            __builtin_prefetch(bp + 32 * DM, 0, 1);
            v16bf b;
#pragma unroll
            for (int s = 0; s < 8; ++s) { b[s] = bp[(size_t)s * DM]; b[s + 8] = bp[(size_t)(16 + s) * DM]; }

            v8bf lo0 = *(const v8bf*)&St[m][f0 + 8 * kh];
            v8bf hi0 = *(const v8bf*)&St[m][f0 + 16 + 8 * kh];
            v16bf a0 = __builtin_shufflevector(lo0, hi0, 0, 1, 2, 3, 4, 5, 6, 7,
                                               8, 9, 10, 11, 12, 13, 14, 15);
            c0 = __builtin_amdgcn_wmma_f32_16x16x32_bf16(false, a0, false, b,
                                                         (short)0, c0, false, false);

            v8bf lo1 = *(const v8bf*)&St[m + 16][f0 + 8 * kh];
            v8bf hi1 = *(const v8bf*)&St[m + 16][f0 + 16 + 8 * kh];
            v16bf a1 = __builtin_shufflevector(lo1, hi1, 0, 1, 2, 3, 4, 5, 6, 7,
                                               8, 9, 10, 11, 12, 13, 14, 15);
            c1 = __builtin_amdgcn_wmma_f32_16x16x32_bf16(false, a1, false, b,
                                                         (short)0, c1, false, false);
        }
        const int d = n0 + m;
#pragma unroll
        for (int r = 0; r < 8; ++r) {
            int M = r + 8 * kh;
            unsafeAtomicAdd(&out[(size_t)toks[M] * DM + d], c0[r]);
            unsafeAtomicAdd(&out[(size_t)toks[M + 16] * DM + d], c1[r]);
        }
    }
}

// ---------------------------------------------------------------- launch
extern "C" void kernel_launch(void* const* d_in, const int* in_sizes, int n_in,
                              void* d_out, int out_size, void* d_ws, size_t ws_size,
                              hipStream_t stream) {
    (void)in_sizes; (void)n_in; (void)out_size; (void)ws_size;
    const float* token_stream    = (const float*)d_in[0];
    const float* selection_input = (const float*)d_in[1];
    const float* keys_w          = (const float*)d_in[2];
    const float* values_w        = (const float*)d_in[3];
    const float* expert_sel      = (const float*)d_in[4];
    const float* bias_ffn        = (const float*)d_in[5];

    float* out     = (float*)d_out;
    int*   sel_out = (int*)((float*)d_out + (size_t)TOK * DM);  // second output

    // workspace carve (256B aligned)
    char* ws = (char*)d_ws;
    size_t o = 0;
    auto carve = [&](size_t bytes) { char* p = ws + o; o = (o + bytes + 255) & ~(size_t)255; return p; };
    __bf16* keysB      = (__bf16*)carve((size_t)NE * DM * FE * 2);
    __bf16* valuesB    = (__bf16*)carve((size_t)NE * FE * DM * 2);
    float*  wdense     = (float*)carve((size_t)TOK * NE * 4);
    int*    pair_token = (int*)carve((size_t)TOK * KSEL * 4);
    int*    counts     = (int*)carve(NE * 4);
    int*    offsets    = (int*)carve(NE * 4);
    int*    cursor     = (int*)carve(NE * 4);

    moe_init<<<2048, 256, 0, stream>>>(out, counts);
    moe_cvt<<<4096, 256, 0, stream>>>(keys_w, values_w, keysB, valuesB);
    moe_route<<<TOK / 8, 256, 0, stream>>>(selection_input, expert_sel, bias_ffn,
                                           wdense, sel_out, counts);
    moe_scan<<<1, 32, 0, stream>>>(counts, offsets, cursor);
    moe_scatter<<<TOK / 256, 256, 0, stream>>>(sel_out, cursor, pair_token);

    dim3 grid(TOK / MT, NE);   // max tiles per expert x experts; tiles early-exit past count
    moe_gemm<<<grid, 256, 0, stream>>>(token_stream, keysB, valuesB, wdense,
                                       pair_token, counts, offsets, out);
}
